// RQKMeans_46600395162149
// MI455X (gfx1250) — matrix operand order repaired and checked by reading, not compiled, not run on hardware
//
#include <hip/hip_runtime.h>

// ---------------------------------------------------------------------------
// Residual VQ (4 levels, K=256, D=128) for MI455X / gfx1250.
// argmin via bf16 WMMA with hi/lo error compensation (3 wmmas ~= fp32 dot);
// 6 independent accumulator chains (2 tiles x 3 terms) -> no WMMA hazard NOPs.
// gather/update in exact fp32; recon = x - final residual.
// ---------------------------------------------------------------------------

#define DIMF 128
#define KCB 256
#define NLVL 4
#define CBP 280          // LDS row stride (ushorts): [hi 128 | pad 8 | lo 128 | pad 16]
#define LOOFF 136        // lo half offset within a row (ushorts)
#define ROWS_PER_WG 128  // 8 waves * 16 rows

typedef __bf16 v16bf __attribute__((ext_vector_type(16)));
typedef float v8f __attribute__((ext_vector_type(8)));
typedef unsigned short v8u16 __attribute__((ext_vector_type(8)));
typedef unsigned short v16u16 __attribute__((ext_vector_type(16)));

__device__ __forceinline__ unsigned short bf16_rne(float f) {
  unsigned int u = __float_as_uint(f);
  u += 0x7FFFu + ((u >> 16) & 1u);
  return (unsigned short)(u >> 16);
}
__device__ __forceinline__ float bf16_f32(unsigned short h) {
  return __uint_as_float(((unsigned int)h) << 16);
}

// Load 16 floats of one row in the WMMA A-operand striping for this lane:
// elements [base .. base+7] and [base+16 .. base+23], base given in float4 units.
__device__ __forceinline__ void load16(float* dst, const float4* p, int b) {
  float4 v0 = p[b], v1 = p[b + 1], v2 = p[b + 4], v3 = p[b + 5];
  dst[0] = v0.x;  dst[1] = v0.y;  dst[2] = v0.z;  dst[3] = v0.w;
  dst[4] = v1.x;  dst[5] = v1.y;  dst[6] = v1.z;  dst[7] = v1.w;
  dst[8] = v2.x;  dst[9] = v2.y;  dst[10] = v2.z; dst[11] = v2.w;
  dst[12] = v3.x; dst[13] = v3.y; dst[14] = v3.z; dst[15] = v3.w;
}

__device__ __forceinline__ v16bf lds_b16x16(const unsigned short* p) {
  v8u16 a = *(const v8u16*)p;
  v8u16 b = *(const v8u16*)(p + 16);
  return __builtin_bit_cast(
      v16bf, __builtin_shufflevector(a, b, 0, 1, 2, 3, 4, 5, 6, 7, 8, 9, 10,
                                     11, 12, 13, 14, 15));
}

extern "C" __global__ void __launch_bounds__(256, 4)
rq_kernel(const float* __restrict__ x,
          const float* __restrict__ cb,
          float* __restrict__ recon_out,
          int* __restrict__ codes_out) {
  extern __shared__ __align__(16) char smem_raw[];
  unsigned short* cb_sh = (unsigned short*)smem_raw;  // KCB rows of CBP (hi|lo)
  float* c2s = (float*)(cb_sh + KCB * CBP);           // KCB ||c||^2
  int* idx_sh = (int*)(c2s + KCB);                    // 8 waves * 16 rows

  const int tid = threadIdx.x;
  const int wave = tid >> 5;
  const int lane = tid & 31;
  const int r = lane & 15;
  const bool lo_half = lane < 16;
  const int row = blockIdx.x * ROWS_PER_WG + wave * 16 + r;

  // Residual in registers, A-operand striped layout:
  // chunk c (K block of 32): this lane holds K = 32c+off .. +7 and 32c+off+16 .. +23
  // with off = 0 (lanes 0-15) or 8 (lanes 16-31).
  float res[64];
  {
    const float4* xr = (const float4*)(x + (size_t)row * DIMF);
#pragma unroll
    for (int c = 0; c < 4; ++c)
      load16(&res[c * 16], xr, 8 * c + (lo_half ? 0 : 2));
  }

  for (int l = 0; l < NLVL; ++l) {
    // ---- stage codebook level l into LDS as bf16 hi/lo (interleaved) + ||c||^2
    if (tid < KCB) c2s[tid] = 0.0f;
    __syncthreads();
    {
      const float4* src = (const float4*)(cb + (size_t)l * KCB * DIMF);
#pragma unroll 1
      for (int i = 0; i < (KCB * DIMF / 4) / 256; ++i) {  // 32 iters
        int e4 = tid + 256 * i;
        float4 v = src[e4];
        int e = e4 * 4;
        int crow = e >> 7;
        int k = e & 127;
        unsigned short h0 = bf16_rne(v.x), h1 = bf16_rne(v.y);
        unsigned short h2 = bf16_rne(v.z), h3 = bf16_rne(v.w);
        unsigned short q0 = bf16_rne(v.x - bf16_f32(h0));
        unsigned short q1 = bf16_rne(v.y - bf16_f32(h1));
        unsigned short q2 = bf16_rne(v.z - bf16_f32(h2));
        unsigned short q3 = bf16_rne(v.w - bf16_f32(h3));
        uint2 hp, lp;
        hp.x = (unsigned)h0 | ((unsigned)h1 << 16);
        hp.y = (unsigned)h2 | ((unsigned)h3 << 16);
        lp.x = (unsigned)q0 | ((unsigned)q1 << 16);
        lp.y = (unsigned)q2 | ((unsigned)q3 << 16);
        *(uint2*)(cb_sh + crow * CBP + k) = hp;
        *(uint2*)(cb_sh + crow * CBP + LOOFF + k) = lp;
        atomicAdd(&c2s[crow], v.x * v.x + v.y * v.y + v.z * v.z + v.w * v.w);
      }
    }
    __syncthreads();

    // ---- residual -> bf16 hi/lo A operands ----
    v16bf a_hi[4], a_lo[4];
#pragma unroll
    for (int c = 0; c < 4; ++c) {
      v16u16 ah, al;
#pragma unroll
      for (int j = 0; j < 16; ++j) {
        float v = res[c * 16 + j];
        unsigned short h = bf16_rne(v);
        ah[j] = h;
        al[j] = bf16_rne(v - bf16_f32(h));
      }
      a_hi[c] = __builtin_bit_cast(v16bf, ah);
      a_lo[c] = __builtin_bit_cast(v16bf, al);
    }

    // ---- scan 256 codes as 8 pairs of 16-wide N-tiles.
    // C/D layout: VGPR v -> row (v + 8*(lane>=16)), lane%16 -> col within tile.
    float bs[8];
    int bi[8];
#pragma unroll
    for (int v = 0; v < 8; ++v) { bs[v] = -3.4e38f; bi[v] = 0; }
    const int off_b = lo_half ? 0 : 8;  // B layout: low lanes K 0-7,16-23, high lanes K 8-15,24-31 per chunk

#pragma unroll 1
    for (int tp = 0; tp < 8; ++tp) {
      const int n0 = (2 * tp) * 16 + r;
      const int n1 = (2 * tp + 1) * 16 + r;
      const unsigned short* p0 = cb_sh + n0 * CBP;
      const unsigned short* p1 = cb_sh + n1 * CBP;
      v8f hh0 = {0, 0, 0, 0, 0, 0, 0, 0}, hh1 = {0, 0, 0, 0, 0, 0, 0, 0};
      v8f lh0 = {0, 0, 0, 0, 0, 0, 0, 0}, lh1 = {0, 0, 0, 0, 0, 0, 0, 0};
      v8f hl0 = {0, 0, 0, 0, 0, 0, 0, 0}, hl1 = {0, 0, 0, 0, 0, 0, 0, 0};
#pragma unroll
      for (int c = 0; c < 4; ++c) {
        const int ko = 32 * c + off_b;
        v16bf bh0 = lds_b16x16(p0 + ko);
        v16bf bl0 = lds_b16x16(p0 + LOOFF + ko);
        v16bf bh1 = lds_b16x16(p1 + ko);
        v16bf bl1 = lds_b16x16(p1 + LOOFF + ko);
        // round-robin across 6 chains -> dependency distance 6 (>= 5, no NOPs)
        hh0 = __builtin_amdgcn_wmma_f32_16x16x32_bf16(false, a_hi[c], false, bh0,
                                                      (short)0, hh0, false, false);
        hh1 = __builtin_amdgcn_wmma_f32_16x16x32_bf16(false, a_hi[c], false, bh1,
                                                      (short)0, hh1, false, false);
        lh0 = __builtin_amdgcn_wmma_f32_16x16x32_bf16(false, a_lo[c], false, bh0,
                                                      (short)0, lh0, false, false);
        lh1 = __builtin_amdgcn_wmma_f32_16x16x32_bf16(false, a_lo[c], false, bh1,
                                                      (short)0, lh1, false, false);
        hl0 = __builtin_amdgcn_wmma_f32_16x16x32_bf16(false, a_hi[c], false, bl0,
                                                      (short)0, hl0, false, false);
        hl1 = __builtin_amdgcn_wmma_f32_16x16x32_bf16(false, a_hi[c], false, bl1,
                                                      (short)0, hl1, false, false);
      }
      const float hc0 = 0.5f * c2s[n0];
      const float hc1 = 0.5f * c2s[n1];
#pragma unroll
      for (int v = 0; v < 8; ++v) {  // keep ascending-n order for tie-breaking
        float s0 = (hh0[v] + lh0[v] + hl0[v]) - hc0;
        if (s0 > bs[v]) { bs[v] = s0; bi[v] = n0; }
        float s1 = (hh1[v] + lh1[v] + hl1[v]) - hc1;
        if (s1 > bs[v]) { bs[v] = s1; bi[v] = n1; }
      }
    }

    // ---- argmax reduce across the 16 lanes of each half (xor stays in half) ----
#pragma unroll
    for (int off = 8; off >= 1; off >>= 1) {
#pragma unroll
      for (int v = 0; v < 8; ++v) {
        float os = __shfl_xor(bs[v], off, 32);
        int oi = __shfl_xor(bi[v], off, 32);
        if (os > bs[v] || (os == bs[v] && oi < bi[v])) { bs[v] = os; bi[v] = oi; }
      }
    }

    // ---- publish per-row winners via LDS (wave-local; LDS is in-order per wave)
    int* my_idx = idx_sh + wave * 16;
    if (lane == 0) {
#pragma unroll
      for (int v = 0; v < 8; ++v) my_idx[v] = bi[v];
    }
    if (lane == 16) {
#pragma unroll
      for (int v = 0; v < 8; ++v) my_idx[8 + v] = bi[v];
    }
    int code = my_idx[r];
    if (lo_half) codes_out[(size_t)row * NLVL + l] = code;

    // ---- exact fp32 gather from L2-resident codebook; update residual ----
    {
      const float4* s4 = (const float4*)(cb + ((size_t)l * KCB + code) * DIMF);
#pragma unroll
      for (int c = 0; c < 4; ++c) {
        float sel[16];
        load16(sel, s4, 8 * c + (lo_half ? 0 : 2));
#pragma unroll
        for (int j = 0; j < 16; ++j) res[c * 16 + j] -= sel[j];
      }
    }
    __syncthreads();  // LDS codebook reusable for next level
  }

  // ---- recon = x - final residual (reload x; row fully covered by lanes m, m+16)
  {
    const float4* xr = (const float4*)(x + (size_t)row * DIMF);
    float4* o = (float4*)(recon_out + (size_t)row * DIMF);
#pragma unroll
    for (int c = 0; c < 4; ++c) {
      int b = 8 * c + (lo_half ? 0 : 2);
      float xv[16];
      load16(xv, xr, b);
      o[b] = make_float4(xv[0] - res[c * 16 + 0], xv[1] - res[c * 16 + 1],
                         xv[2] - res[c * 16 + 2], xv[3] - res[c * 16 + 3]);
      o[b + 1] = make_float4(xv[4] - res[c * 16 + 4], xv[5] - res[c * 16 + 5],
                             xv[6] - res[c * 16 + 6], xv[7] - res[c * 16 + 7]);
      o[b + 4] = make_float4(xv[8] - res[c * 16 + 8], xv[9] - res[c * 16 + 9],
                             xv[10] - res[c * 16 + 10], xv[11] - res[c * 16 + 11]);
      o[b + 5] = make_float4(xv[12] - res[c * 16 + 12], xv[13] - res[c * 16 + 13],
                             xv[14] - res[c * 16 + 14], xv[15] - res[c * 16 + 15]);
    }
  }
}

extern "C" void kernel_launch(void* const* d_in, const int* in_sizes, int n_in,
                              void* d_out, int out_size, void* d_ws, size_t ws_size,
                              hipStream_t stream) {
  (void)n_in; (void)out_size; (void)d_ws; (void)ws_size;
  const float* x = (const float*)d_in[0];
  const float* cb = (const float*)d_in[1];
  const int rows = in_sizes[0] / DIMF;  // 262144

  float* recon = (float*)d_out;
  int* codes = (int*)(recon + (size_t)rows * DIMF);

  const size_t smem = (size_t)KCB * CBP * 2   // interleaved hi/lo codebook (bf16)
                    + (size_t)KCB * 4         // c2
                    + 8 * 16 * 4;             // idx_sh
  (void)hipFuncSetAttribute((const void*)rq_kernel,
                            hipFuncAttributeMaxDynamicSharedMemorySize,
                            (int)smem);
  rq_kernel<<<rows / ROWS_PER_WG, 256, smem, stream>>>(x, cb, recon, codes);
}
